// PamirNetMultiview_20564303413292
// MI455X (gfx1250) — matrix operand order repaired and checked by reading, not compiled, not run on hardware
//
#include <hip/hip_runtime.h>
#include <hip/hip_bf16.h>
#include <math.h>

// ---------------------------------------------------------------------------
// PamirNet multiview, fused for MI455X (gfx1250, wave32, WMMA bf16 16x16x32)
// ---------------------------------------------------------------------------
typedef __attribute__((ext_vector_type(16))) __bf16 v16bf;
typedef __attribute__((ext_vector_type(8)))  float  v8f;

#define WAVES 8
#define NT    64                 // points per block (4 x 16-point WMMA tiles)

// LDS layout (units: __bf16 halves unless noted)
// Activations stored in WMMA D-register order per 16x16 tile:
//   element (k,p): k16=k/16, r=k%16, col=p%16
//   addr = ntile*STRIDE + (k16*32 + col + 16*(r>=8 ? 1:0))*8 + (r&7)
#define X_K16       18                       // 288 / 16
#define X_NTSTRIDE  (X_K16 * 256)            // 4608 halves per 16-pt tile
#define X_OFF       0
#define A_NTSTRIDE  (64 * 256)               // out0: 1024 ch -> 64 k16 tiles
#define A_OFF       (4 * X_NTSTRIDE)         // 18432
#define B_NTSTRIDE  (32 * 256)               // out1: 512 ch
#define B_OFF       (A_OFF + 4 * A_NTSTRIDE) // 83968
#define HALVES_TOT  (B_OFF + 4 * B_NTSTRIDE) // 116736 halves = 233472 B
#define S_OFF_BYTES (HALVES_TOT * 2)
#define S_FLOATS    (4 * 8 * 256)            // out3 view-sum: 128 ch x 64 pts
#define LDS_BYTES   (S_OFF_BYTES + S_FLOATS * 4)   // 266240 B (< 320KB WGP LDS)

// weight regions in workspace (halves)
#define W0_OFF 0
#define W1_OFF 294912       // 1024*288
#define W2_OFF 966656       // +512*1312
#define W3_OFF 1171456      // +256*800
#define W_TOTAL_HALVES 1241088

// ---------------------------------------------------------------------------
// Weight pack: fp32 row-major [O][K] -> bf16 A-tile swizzle.
// A 16x32 tile, lane L: row M=L%16, hi=L/16, half j: K = hi*16 + j
// Tile storage: tile(mt,kt) base = (mt*Ktiles+kt)*512, lane chunk = lane*16+j.
// ---------------------------------------------------------------------------
__global__ void pack_weights(const float* __restrict__ W, __bf16* __restrict__ dst,
                             int O, int K)
{
    int idx = blockIdx.x * blockDim.x + threadIdx.x;
    if (idx >= O * K) return;
    int o = idx / K, k = idx % K;
    int mt = o >> 4, r = o & 15;
    int kt = k >> 5, kk = k & 31;
    int hi   = kk >> 4;
    int j    = kk & 15;
    int lane = r + (hi << 4);
    int Kt   = K >> 5;
    dst[((size_t)(mt * Kt + kt) * 32 + lane) * 16 + j] = (__bf16)W[(size_t)o * K + k];
}

// ---------------------------------------------------------------------------
// store a sampled activation element into the swizzled LDS x-buffer
// ---------------------------------------------------------------------------
__device__ __forceinline__ void store_x(__bf16* xb, int pi, int k, float v)
{
    int nt = pi >> 4, col = pi & 15;
    int k16 = k >> 4, r = k & 15;
    xb[nt * X_NTSTRIDE + (k16 * 32 + col + ((r >> 3) << 4)) * 8 + (r & 7)] = (__bf16)v;
}

// ---------------------------------------------------------------------------
// One GEMM layer: D(O x 64) = lrelu(Wsw(O x Kin) * act(Kin x 64) + bias)
// Kin = 288 (x region) + prev-layer channels; kt<9 -> x region, else prev.
// B operand (32x16 bf16, lane=column, K-half by lane/16): assembled from two
// ds_load_b128 of the D-swizzled activation store.
// ---------------------------------------------------------------------------
template<int MT, int KT, int PREVSTRIDE, int DSTSTRIDE, bool LAST>
__device__ __forceinline__ void gemm_layer(
    const __bf16* __restrict__ wsw, const float* __restrict__ bias,
    const __bf16* __restrict__ xb,  const __bf16* __restrict__ prev,
    __bf16* __restrict__ dst, float* __restrict__ accbuf,
    int wave, int lane)
{
    const int col   = lane & 15;
    const int khalf = lane >> 4;

    for (int mt = wave; mt < MT; mt += WAVES) {
        v8f acc[4];
        #pragma unroll
        for (int nt = 0; nt < 4; ++nt) acc[nt] = (v8f){0,0,0,0,0,0,0,0};

        const __bf16* wtile = wsw + (size_t)mt * KT * 512 + lane * 16;

        for (int kt = 0; kt < KT; ++kt) {
            v16bf a = *(const v16bf*)(wtile + kt * 512);
            if (kt + 1 < KT)  // WGP-scope prefetch: pull next A tile into near cache
                __builtin_prefetch(wtile + (kt + 1) * 512, 0, 3);

            #pragma unroll
            for (int nt = 0; nt < 4; ++nt) {
                const __bf16* s;
                if (kt < 9)
                    s = xb   + nt * X_NTSTRIDE + ((kt * 2 + khalf) * 32 + col) * 8;
                else
                    s = prev + nt * PREVSTRIDE + (((kt - 9) * 2 + khalf) * 32 + col) * 8;
                union { v16bf v; uint4 q[2]; } u;
                u.q[0] = *(const uint4*)(s);        // rows 0-7 of this k16 tile
                u.q[1] = *(const uint4*)(s + 128);  // rows 8-15 (lane col+16 chunk)
                acc[nt] = __builtin_amdgcn_wmma_f32_16x16x32_bf16(
                    false, a, false, u.v, (short)0, acc[nt], false, false);
            }
        }

        float bv[8];
        #pragma unroll
        for (int v = 0; v < 8; ++v) bv[v] = bias[mt * 16 + khalf * 8 + v];

        #pragma unroll
        for (int nt = 0; nt < 4; ++nt) {
            if (LAST) {
                float* ab = accbuf + nt * 2048 + (mt * 32 + lane) * 8;
                #pragma unroll
                for (int v = 0; v < 8; ++v) {
                    float x = acc[nt][v] + bv[v];
                    ab[v] += (x >= 0.f) ? x : 0.2f * x;
                }
            } else {
                union { __bf16 h[8]; uint4 q; } o;
                #pragma unroll
                for (int v = 0; v < 8; ++v) {
                    float x = acc[nt][v] + bv[v];
                    o.h[v] = (__bf16)((x >= 0.f) ? x : 0.2f * x);
                }
                *(uint4*)(dst + nt * DSTSTRIDE + (mt * 32 + lane) * 8) = o.q;
            }
        }
    }
}

// ---------------------------------------------------------------------------
// Fused kernel: grid = (512 point tiles, 2 levels), 256 threads, 260KB LDS.
// ---------------------------------------------------------------------------
__global__ __launch_bounds__(256)
void pamir_fused(const float* __restrict__ pts, const float* __restrict__ pts_proj,
                 const float* __restrict__ img0, const float* __restrict__ img1,
                 const float* __restrict__ vol0, const float* __restrict__ vol1,
                 const __bf16* __restrict__ wall,
                 const float* __restrict__ b0, const float* __restrict__ b1,
                 const float* __restrict__ b2, const float* __restrict__ b3,
                 const float* __restrict__ W4, const float* __restrict__ b4,
                 const int* __restrict__ mean_num,
                 float* __restrict__ out)
{
    extern __shared__ char smem[];
    __bf16* xb   = (__bf16*)smem;
    __bf16* bufA = (__bf16*)smem + A_OFF;
    __bf16* bufB = (__bf16*)smem + B_OFF;
    float*  sbuf = (float*)(smem + S_OFF_BYTES);

    const int tid  = threadIdx.x;
    const int wave = tid >> 5, lane = tid & 31;
    const int tileBase = blockIdx.x * NT;
    const int level    = blockIdx.y;
    const float* img = level ? img1 : img0;
    const float* vol = level ? vol1 : vol0;

    const __bf16* w0 = wall + W0_OFF;
    const __bf16* w1 = wall + W1_OFF;
    const __bf16* w2 = wall + W2_OFF;
    const __bf16* w3 = wall + W3_OFF;

    for (int i = tid; i < S_FLOATS; i += 256) sbuf[i] = 0.f;

    int mn = mean_num[0];
    if (mn > 3) mn = 3;
    if (mn < 1) mn = 1;

    // ---- per-thread sampling coordinates (4 threads per point) ----
    const int pi = tid >> 2, q = tid & 3;
    const int pg = tileBase + pi;

    float gx = pts_proj[pg * 2 + 0], gy = pts_proj[pg * 2 + 1];
    float ix = fminf(fmaxf(((gx + 1.f) * 128.f - 1.f) * 0.5f, 0.f), 127.f);
    float iy = fminf(fmaxf(((gy + 1.f) * 128.f - 1.f) * 0.5f, 0.f), 127.f);
    int   x0 = (int)ix, y0 = (int)iy;
    float wx = ix - (float)x0, wy = iy - (float)y0;
    int   x1 = min(x0 + 1, 127), y1 = min(y0 + 1, 127);

    float px = pts[pg * 3 + 0] * 2.f;
    float py = pts[pg * 3 + 1] * 2.f;
    float pz = pts[pg * 3 + 2] * 2.f;
    float jx = fminf(fmaxf(((px + 1.f) * 32.f - 1.f) * 0.5f, 0.f), 31.f);
    float jy = fminf(fmaxf(((py + 1.f) * 32.f - 1.f) * 0.5f, 0.f), 31.f);
    float jz = fminf(fmaxf(((pz + 1.f) * 32.f - 1.f) * 0.5f, 0.f), 31.f);
    int   a0 = (int)jx, c0 = (int)jy, e0 = (int)jz;
    float ux = jx - (float)a0, uy = jy - (float)c0, uz = jz - (float)e0;
    int   a1 = min(a0 + 1, 31), c1 = min(c0 + 1, 31), e1 = min(e0 + 1, 31);

    for (int view = 0; view < mn; ++view) {
        __syncthreads();   // xb free for reuse; also covers sbuf zeroing (view 0)

        // ---- bilinear sample: 256 image channels (64 per thread) ----
        const float* ib = img + (size_t)view * 256 * 16384;
        for (int i = 0; i < 64; ++i) {
            int c = (q << 6) + i;
            const float* pl = ib + (size_t)c * 16384;
            float v00 = pl[y0 * 128 + x0], v01 = pl[y0 * 128 + x1];
            float v10 = pl[y1 * 128 + x0], v11 = pl[y1 * 128 + x1];
            float val = v00 * (1.f - wy) * (1.f - wx) + v01 * (1.f - wy) * wx
                      + v10 * wy * (1.f - wx)         + v11 * wy * wx;
            store_x(xb, pi, c, val);
        }
        // ---- trilinear sample: 32 volume channels (8 per thread) ----
        const float* vb = vol + (size_t)view * 32 * 32768;
        for (int i = 0; i < 8; ++i) {
            int c = (q << 3) + i;
            const float* pl = vb + (size_t)c * 32768;
            float c000 = pl[e0 * 1024 + c0 * 32 + a0];
            float c001 = pl[e0 * 1024 + c0 * 32 + a1];
            float c010 = pl[e0 * 1024 + c1 * 32 + a0];
            float c011 = pl[e0 * 1024 + c1 * 32 + a1];
            float c100 = pl[e1 * 1024 + c0 * 32 + a0];
            float c101 = pl[e1 * 1024 + c0 * 32 + a1];
            float c110 = pl[e1 * 1024 + c1 * 32 + a0];
            float c111 = pl[e1 * 1024 + c1 * 32 + a1];
            float val = c000 * (1.f-uz)*(1.f-uy)*(1.f-ux) + c001 * (1.f-uz)*(1.f-uy)*ux
                      + c010 * (1.f-uz)*uy*(1.f-ux)       + c011 * (1.f-uz)*uy*ux
                      + c100 * uz*(1.f-uy)*(1.f-ux)       + c101 * uz*(1.f-uy)*ux
                      + c110 * uz*uy*(1.f-ux)             + c111 * uz*uy*ux;
            store_x(xb, pi, 256 + c, val);
        }
        __syncthreads();

        // ---- MLP with dense skip concats (x routed by K-tile index) ----
        gemm_layer<64,  9, X_NTSTRIDE, A_NTSTRIDE, false>(w0, b0, xb, xb,   bufA, nullptr, wave, lane);
        __syncthreads();
        gemm_layer<32, 41, A_NTSTRIDE, B_NTSTRIDE, false>(w1, b1, xb, bufA, bufB, nullptr, wave, lane);
        __syncthreads();
        gemm_layer<16, 25, B_NTSTRIDE, 4096,       false>(w2, b2, xb, bufB, bufA, nullptr, wave, lane);
        __syncthreads();
        gemm_layer< 8, 17, 4096,       0,          true >(w3, b3, xb, bufA, nullptr, sbuf, wave, lane);
    }
    __syncthreads();

    // ---- head: mean over views, 1x128 conv, sigmoid ----
    if (tid < NT) {
        int p = tid, ppg = tileBase + p;
        int ntp = p >> 4, colp = p & 15;
        float acc = 0.f;
        for (int c = 0; c < 128; ++c) {
            int k16 = c >> 4, r = c & 15;
            acc += sbuf[ntp * 2048 + (k16 * 32 + colp + ((r >> 3) << 4)) * 8 + (r & 7)] * W4[c];
        }
        float val = acc / (float)mn + b4[0];
        out[level * 32768 + ppg] = 1.f / (1.f + expf(-val));
    }
}

// ---------------------------------------------------------------------------
extern "C" void kernel_launch(void* const* d_in, const int* in_sizes, int n_in,
                              void* d_out, int out_size, void* d_ws, size_t ws_size,
                              hipStream_t stream)
{
    const float* pts      = (const float*)d_in[0];
    const float* pts_proj = (const float*)d_in[1];
    const float* img0     = (const float*)d_in[2];
    const float* img1     = (const float*)d_in[3];
    const float* vol0     = (const float*)d_in[4];
    const float* vol1     = (const float*)d_in[5];
    const float* W0       = (const float*)d_in[6];
    const float* b0       = (const float*)d_in[7];
    const float* W1       = (const float*)d_in[8];
    const float* b1       = (const float*)d_in[9];
    const float* W2       = (const float*)d_in[10];
    const float* b2       = (const float*)d_in[11];
    const float* W3       = (const float*)d_in[12];
    const float* b3       = (const float*)d_in[13];
    const float* W4       = (const float*)d_in[14];
    const float* b4       = (const float*)d_in[15];
    const int*   mean_num = (const int*)d_in[16];
    float*       out      = (float*)d_out;

    __bf16* wsw = (__bf16*)d_ws;

    // pack weights fp32 -> bf16, WMMA A-tile swizzle (deterministic, every call)
    {
        int n;
        n = 1024 * 288;  pack_weights<<<(n + 255) / 256, 256, 0, stream>>>(W0, wsw + W0_OFF, 1024, 288);
        n = 512 * 1312;  pack_weights<<<(n + 255) / 256, 256, 0, stream>>>(W1, wsw + W1_OFF, 512, 1312);
        n = 256 * 800;   pack_weights<<<(n + 255) / 256, 256, 0, stream>>>(W2, wsw + W2_OFF, 256, 800);
        n = 128 * 544;   pack_weights<<<(n + 255) / 256, 256, 0, stream>>>(W3, wsw + W3_OFF, 128, 544);
    }

    (void)hipFuncSetAttribute((const void*)pamir_fused,
                              hipFuncAttributeMaxDynamicSharedMemorySize, LDS_BYTES);

    dim3 grid(512, 2);  // 512 tiles of 64 points x 2 levels
    pamir_fused<<<grid, 256, LDS_BYTES, stream>>>(
        pts, pts_proj, img0, img1, vol0, vol1, wsw,
        b0, b1, b2, b3, W4, b4, mean_num, out);
}
